// RegisterStep_71347996721406
// MI455X (gfx1250) — matrix operand order repaired and checked by reading, not compiled, not run on hardware
//
#include <hip/hip_runtime.h>
#include <hip/hip_bf16.h>
#include <math.h>
#include <stdint.h>

// ---------------- types ----------------
typedef __attribute__((ext_vector_type(16))) _Float16 v16h;
typedef __attribute__((ext_vector_type(8)))  _Float16 v8h;
typedef __attribute__((ext_vector_type(8)))  float    v8f;
typedef __attribute__((ext_vector_type(4)))  int      v4i;

#define B_  4
#define T_  4096
#define V_  512
#define C_  512
#define NB2 128
#define M_  (B_ * T_)          // 16384 rows
#define EPSF 1.1920929e-07f

#if defined(__gfx1250__) && __has_builtin(__builtin_amdgcn_global_load_async_to_lds_b128)
#define USE_ASYNC 1
typedef __attribute__((address_space(1))) v4i as1_v4i;
typedef __attribute__((address_space(3))) v4i as3_v4i;
#else
#define USE_ASYNC 0
#endif

static __device__ __forceinline__ v8f wmma16(const v16h& a, const v16h& b, const v8f& c) {
    return __builtin_amdgcn_wmma_f32_16x16x32_f16(false, a, false, b, (short)0, c, false, false);
}

// B-operand chunk: 16 contiguous halves at p (two b128 loads; works for global or LDS)
static __device__ __forceinline__ v16h load_b16(const _Float16* p) {
    const v8h* bp = (const v8h*)p;
    v8h b0 = bp[0], b1 = bp[1];
    v16h r;
#pragma unroll
    for (int i = 0; i < 8; ++i) { r[i] = b0[i]; r[8 + i] = b1[i]; }
    return r;
}

// A-operand chunk: halves at p[0..7] and p[16..23]
static __device__ __forceinline__ v16h load_a16(const _Float16* p) {
    const v8h* ap = (const v8h*)p;
    v8h s0 = ap[0], s1 = ap[2];
    v16h r;
#pragma unroll
    for (int i = 0; i < 8; ++i) { r[i] = s0[i]; r[8 + i] = s1[i]; }
    return r;
}

static __device__ __forceinline__ v8f vzero8() {
    v8f z;
#pragma unroll
    for (int i = 0; i < 8; ++i) z[i] = 0.f;
    return z;
}

// ---------------- weight prep: out = basis[v,:] . coeffs[c,:]  (len 128) ----------------
__global__ void proj_kernel(const float* __restrict__ basis, const float* __restrict__ coeffs,
                            _Float16* __restrict__ oh, float* __restrict__ of, int store_cv) {
    int idx = blockIdx.x * blockDim.x + threadIdx.x;       // 512*512
    int a = idx & 511, bq = idx >> 9;
    int v, c;
    if (store_cv) { v = a; c = bq; } else { c = a; v = bq; }
    const float* br = basis + v * NB2;
    const float* cr = coeffs + c * NB2;
    float s = 0.f;
#pragma unroll 8
    for (int j = 0; j < NB2; ++j) s += br[j] * cr[j];
    int di = store_cv ? (c * 512 + v) : (v * 512 + c);
    if (oh) oh[di] = (_Float16)s; else of[di] = s;
}

// softmax over v (axis 0) per column c; rlog/rw stored [c*512+v]
__global__ void softmax_col(const float* __restrict__ rlog, float* __restrict__ rw) {
    __shared__ float red[256];
    int c = blockIdx.x, tid = threadIdx.x;
    const float* row = rlog + c * 512;
    float m = -1e30f;
    for (int v = tid; v < 512; v += 256) m = fmaxf(m, row[v]);
    red[tid] = m; __syncthreads();
    for (int o = 128; o; o >>= 1) { if (tid < o) red[tid] = fmaxf(red[tid], red[tid + o]); __syncthreads(); }
    m = red[0]; __syncthreads();
    float s = 0.f;
    for (int v = tid; v < 512; v += 256) s += expf(row[v] - m);
    red[tid] = s; __syncthreads();
    for (int o = 128; o; o >>= 1) { if (tid < o) red[tid] += red[tid + o]; __syncthreads(); }
    s = red[0];
    float inv = 1.0f / s;
    for (int v = tid; v < 512; v += 256) rw[c * 512 + v] = expf(row[v] - m) * inv;
}

// W_r[v][c] = sum_j read_w[v][j]*cm[j][c]; rw[cc*512+vv] = read_w[vv][cc]
__global__ void mix_kernel(const float* __restrict__ rw, const float* __restrict__ cm,
                           _Float16* __restrict__ wr_t) {
    int idx = blockIdx.x * blockDim.x + threadIdx.x;   // 512*512
    int v = idx & 511, c = idx >> 9;
    float s = 0.f;
#pragma unroll 4
    for (int j = 0; j < 512; ++j) s += rw[j * 512 + v] * cm[j * 512 + c];
    wr_t[c * 512 + v] = (_Float16)s;
}

// per-row inverse RMS
__global__ void row_rsqrt(const float* __restrict__ X, float* __restrict__ rs, int rows) {
    int wave = threadIdx.x >> 5, lane = threadIdx.x & 31;
    int row = blockIdx.x * (blockDim.x >> 5) + wave;
    if (row >= rows) return;
    const float* p = X + (size_t)row * 512;
    float s = 0.f;
#pragma unroll
    for (int i = 0; i < 16; ++i) { float t = p[lane + i * 32]; s += t * t; }
#pragma unroll
    for (int o = 16; o; o >>= 1) s += __shfl_xor(s, o, 32);
    if (lane == 0) rs[row] = rsqrtf(s * (1.0f / 512.0f) + EPSF);
}

// ---------------- generic WMMA GEMM: D = op( A[M,K] * Bt^T ), Bt stored [N,K] f16 ----------------
// MODE 0: out16[m*N+n] = acc
// MODE 1: out16[b*N*Tdim + n*Tdim + t] = acc   (V transposed)
// MODE 2: out32[m*N+n] = addend[m*N+n] + (*scale)*acc
// MODE 3: out16[m*N+n] = gelu(acc + bias[n]) * (*scale)
template <int MODE, bool AF16>
__global__ __launch_bounds__(128)
void gemm16_kernel(const void* __restrict__ Aptr, const float* __restrict__ rowscale,
                   const _Float16* __restrict__ Bt, int N, int K,
                   const float* __restrict__ addend, const float* __restrict__ bias,
                   const float* __restrict__ scale_ptr,
                   float* __restrict__ out32, _Float16* __restrict__ out16, int Tdim) {
    const float*    A32 = (const float*)Aptr;
    const _Float16* A16 = (const _Float16*)Aptr;
    const int wave = threadIdx.x >> 5, lane = threadIdx.x & 31;
    const int lo = lane & 15, hi = lane >> 4;
    const int ntn = N >> 4;
    const int tile = blockIdx.x * 4 + wave;
    const int tm = tile / ntn, tn = tile - tm * ntn;
    const int row = tm * 16 + lo;
    const int coln = tn * 16 + lo;
    const float rsc = AF16 ? 1.0f : rowscale[row];
    const _Float16* brow = Bt + (size_t)coln * K + (hi << 3);

    v8f acc0 = vzero8(), acc1 = vzero8();
    const int KC = K >> 5;   // 16
    for (int kc = 0; kc < KC; kc += 2) {
        __builtin_prefetch((const void*)(brow + (kc << 5) + 1024), 0, 1);
        v16h a0, a1;
        if (AF16) {
            const _Float16* ab = A16 + (size_t)row * K + (hi << 3);
            a0 = load_a16(ab + (kc << 5));
            a1 = load_a16(ab + ((kc + 1) << 5));
        } else {
            const float* ab = A32 + (size_t)row * K + (hi << 3);
#pragma unroll
            for (int u = 0; u < 2; ++u) {
                const float* base = ab + ((kc + u) << 5);
                v16h t;
#pragma unroll
                for (int i = 0; i < 8; ++i) {
                    t[i]     = (_Float16)(base[i] * rsc);
                    t[8 + i] = (_Float16)(base[16 + i] * rsc);
                }
                if (u == 0) a0 = t; else a1 = t;
            }
        }
        v16h b0 = load_a16(brow + (kc << 5));
        v16h b1 = load_a16(brow + ((kc + 1) << 5));
        acc0 = wmma16(a0, b0, acc0);
        acc1 = wmma16(a1, b1, acc1);
    }
    v8f acc = acc0 + acc1;

    const int n = tn * 16 + lo;
#pragma unroll
    for (int r = 0; r < 8; ++r) {
        int m = tm * 16 + r + (hi << 3);
        float v = acc[r];
        if (MODE == 0) {
            out16[(size_t)m * N + n] = (_Float16)v;
        } else if (MODE == 1) {
            int bb = m / Tdim, t = m - bb * Tdim;
            out16[(size_t)bb * N * Tdim + (size_t)n * Tdim + t] = (_Float16)v;
        } else if (MODE == 2) {
            size_t idx = (size_t)m * N + n;
            out32[idx] = addend[idx] + scale_ptr[0] * v;
        } else {
            float u = v + bias[n];
            float g = 0.5f * u * (1.0f + erff(u * 0.70710678118654752f));
            out16[(size_t)m * N + n] = (_Float16)(g * scale_ptr[0]);
        }
    }
}

// ---------------- fused decay attention (4 waves per 16-row t-tile) ----------------
// retrieved[t,:] = sum_{s>t} (q[t].k[s]) * decay^(s-t-1) * v[s,:]
// Wave w owns K-chunks [4w,4w+4) for the score partials and C columns [128w,128w+128) for P@V.
__global__ __launch_bounds__(128)
void attn_kernel(const _Float16* __restrict__ q16,   // [B,T,C]
                 const _Float16* __restrict__ k16,   // [B,T,C]
                 const _Float16* __restrict__ vT,    // [B,C,T]
                 _Float16* __restrict__ ret16,       // [B,T,C] (pre-scaled by mem_out_scale)
                 const float* __restrict__ dlp, const float* __restrict__ mosp) {
    const int tid = threadIdx.x;
    const int wid = tid >> 5, lane = tid & 31;
    const int lo = lane & 15, hi = lane >> 4;
    const int t0 = blockIdx.x * 16;
    const int b  = blockIdx.y;
    const float decay = 1.0f / (1.0f + expf(-dlp[0]));
    const float l2d = log2f(decay);

    __shared__ float    pf32[4][16 * 32];   // 8 KB partial score tiles
    __shared__ _Float16 pSh[16 * 32];       // 1 KB weighted P (A-operand staging)
#if USE_ASYNC
    __shared__ _Float16 kSh[32 * C_];       // 32 KB staged K tile
#endif

    // Q quarter: chunks kc = 4*wid + j
    v16h qa[4];
#pragma unroll
    for (int j = 0; j < 4; ++j)
        qa[j] = load_a16(q16 + (size_t)(b * T_ + t0 + lo) * C_ + (((wid << 2) + j) << 5) + (hi << 3));

    v8f acc[8];
#pragma unroll
    for (int nt = 0; nt < 8; ++nt) acc[nt] = vzero8();

    const int sbeg = t0 & ~31;

#if USE_ASYNC
    // each wave stages bytes [256w,256w+256) of every K row via async b128 copies
    auto issue_tile = [&](int s0) {
        const _Float16* g = k16 + (size_t)(b * T_ + s0 + lane) * C_;
        _Float16* l = kSh + lane * C_;
#pragma unroll
        for (int i = 0; i < 16; ++i) {
            int ch = (wid << 4) + i;
            __builtin_amdgcn_global_load_async_to_lds_b128(
                (as1_v4i*)(unsigned long long)(uintptr_t)(g + ch * 8),
                (as3_v4i*)(unsigned int)(uintptr_t)(l + ch * 8), 0, 0);
        }
    };
    issue_tile(sbeg);
#endif

    for (int sb = sbeg; sb < T_; sb += 32) {
#if USE_ASYNC
#if __has_builtin(__builtin_amdgcn_s_wait_asynccnt)
        __builtin_amdgcn_s_wait_asynccnt(0);
#else
        asm volatile("s_wait_asynccnt 0" ::: "memory");
#endif
#endif
        __syncthreads();   // staged K tile visible / pf32+pSh free for reuse

        // partial scores over this wave's 4 K-chunks, two 16-wide s groups:
        // preload B-operands first, then issue the WMMA chain
#pragma unroll
        for (int g = 0; g < 2; ++g) {
#if USE_ASYNC
            const _Float16* kb_base = kSh + (size_t)((g << 4) + lo) * C_ + (hi << 3);
#else
            const _Float16* kb_base = k16 + (size_t)(b * T_ + sb + (g << 4) + lo) * C_ + (hi << 3);
#endif
            v16h kb[4];
#pragma unroll
            for (int j = 0; j < 4; ++j)
                kb[j] = load_a16(kb_base + (((wid << 2) + j) << 5));
            v8f p = vzero8();
#pragma unroll
            for (int j = 0; j < 4; ++j)
                p = wmma16(qa[j], kb[j], p);
#pragma unroll
            for (int r = 0; r < 8; ++r)
                pf32[wid][(r + (hi << 3)) * 32 + (g << 4) + lo] = p[r];
        }
        __syncthreads();   // all partials in LDS; kSh reads finished

#if USE_ASYNC
        if (sb + 32 < T_) issue_tile(sb + 32);   // overlap next K copy with weighting + P@V
#endif

        // hoist V-tile loads: independent of pSh, overlap with weighting below
        const _Float16* vbase = vT + (size_t)b * C_ * T_ + (size_t)((wid << 7) + lo) * T_ + sb + (hi << 3);
        v16h vb[8];
#pragma unroll
        for (int nt = 0; nt < 8; ++nt)
            vb[nt] = load_a16(vbase + (size_t)(nt << 4) * T_);

        // reduce 4 partials + decay weight -> pSh (128 threads x 4 elements)
        {
            const int idx0 = tid << 2;
            const int m = idx0 >> 5;
            const int t = t0 + m;
#pragma unroll
            for (int j = 0; j < 4; ++j) {
                int idx = idx0 + j;
                int s = sb + (idx & 31);
                float sum = pf32[0][idx] + pf32[1][idx] + pf32[2][idx] + pf32[3][idx];
                int e = s - t - 1;
                float w = (e >= 0) ? exp2f(l2d * (float)e) : 0.0f;
                pSh[idx] = (_Float16)(sum * w);
            }
        }
        __syncthreads();

        // reload full P as A-operand (16x32)
        v16h pa;
#pragma unroll
        for (int i = 0; i < 8; ++i) {
            pa[i]     = pSh[lo * 32 + (hi << 3) + i];
            pa[8 + i] = pSh[lo * 32 + 16 + (hi << 3) + i];
        }
        // retrieved += P @ V: 8 independent back-to-back WMMAs on preloaded operands
#pragma unroll
        for (int nt = 0; nt < 8; ++nt)
            acc[nt] = wmma16(pa, vb[nt], acc[nt]);
        __syncthreads();   // P reads done before next iteration overwrites pf32/pSh
    }

    const float mos = mosp[0];
#pragma unroll
    for (int nt = 0; nt < 8; ++nt)
#pragma unroll
        for (int r = 0; r < 8; ++r)
            ret16[(size_t)(b * T_ + t0 + r + (hi << 3)) * C_ + (wid << 7) + (nt << 4) + lo] =
                (_Float16)(acc[nt][r] * mos);
}

// ---------------- host launch ----------------
extern "C" void kernel_launch(void* const* d_in, const int* in_sizes, int n_in,
                              void* d_out, int out_size, void* d_ws, size_t ws_size,
                              hipStream_t stream) {
    (void)in_sizes; (void)n_in; (void)out_size; (void)ws_size;
    const float* x        = (const float*)d_in[0];
    const float* basis    = (const float*)d_in[1];
    const float* q_coeff  = (const float*)d_in[2];
    const float* k_coeff  = (const float*)d_in[3];
    const float* v_coeff  = (const float*)d_in[4];
    const float* o_coeff  = (const float*)d_in[5];
    const float* decay_l  = (const float*)d_in[6];
    const float* mem_os   = (const float*)d_in[7];
    const float* r_coeff  = (const float*)d_in[8];
    const float* w_coeff  = (const float*)d_in[9];
    const float* ch_mix   = (const float*)d_in[10];
    const float* bias     = (const float*)d_in[11];
    const float* op_os    = (const float*)d_in[12];
    const float* mem_sc   = (const float*)d_in[13];
    const float* op_sc    = (const float*)d_in[14];
    float* out = (float*)d_out;

    size_t off = 0;
    auto carve = [&](size_t bytes) {
        void* p = (char*)d_ws + off;
        off += (bytes + 255) & ~(size_t)255;
        return p;
    };
    _Float16* qw_t  = (_Float16*)carve(512 * 512 * 2);   // [C,V]
    _Float16* kw_t  = (_Float16*)carve(512 * 512 * 2);   // [C,V]
    _Float16* vw_t  = (_Float16*)carve(512 * 512 * 2);   // [C,V]
    _Float16* ow_t  = (_Float16*)carve(512 * 512 * 2);   // [V,C]
    _Float16* ww_t  = (_Float16*)carve(512 * 512 * 2);   // [V,C]
    _Float16* wr_t  = (_Float16*)carve(512 * 512 * 2);   // [C,V]
    float*    rlog  = (float*)carve(512 * 512 * 4);
    float*    rw    = (float*)carve(512 * 512 * 4);
    float*    rs_x  = (float*)carve(M_ * 4);
    float*    rs_x1 = (float*)carve(M_ * 4);
    _Float16* q16   = (_Float16*)carve((size_t)M_ * C_ * 2);
    _Float16* k16   = (_Float16*)carve((size_t)M_ * C_ * 2);
    _Float16* vT16  = (_Float16*)carve((size_t)M_ * C_ * 2);
    _Float16* ret16 = (_Float16*)carve((size_t)M_ * C_ * 2);
    float*    x1    = (float*)carve((size_t)M_ * V_ * 4);
    _Float16* vals16 = q16; // reuse: q16 dead after attention

    // weight prep
    proj_kernel<<<1024, 256, 0, stream>>>(basis, q_coeff, qw_t, nullptr, 1);
    proj_kernel<<<1024, 256, 0, stream>>>(basis, k_coeff, kw_t, nullptr, 1);
    proj_kernel<<<1024, 256, 0, stream>>>(basis, v_coeff, vw_t, nullptr, 1);
    proj_kernel<<<1024, 256, 0, stream>>>(basis, o_coeff, ow_t, nullptr, 0);
    proj_kernel<<<1024, 256, 0, stream>>>(basis, w_coeff, ww_t, nullptr, 0);
    proj_kernel<<<1024, 256, 0, stream>>>(basis, r_coeff, nullptr, rlog, 1);
    softmax_col<<<512, 256, 0, stream>>>(rlog, rw);
    mix_kernel<<<1024, 256, 0, stream>>>(rw, ch_mix, wr_t);

    // stage 1: rms-norm + q/k/v projections (v stored transposed [B,C,T])
    row_rsqrt<<<M_ / 8, 256, 0, stream>>>(x, rs_x, M_);
    const int GB = (M_ / 16) * (V_ / 16) / 4; // 8192 blocks, 4 tiles each
    gemm16_kernel<0, false><<<GB, 128, 0, stream>>>(x, rs_x, qw_t, C_, V_,
                                                    nullptr, nullptr, nullptr, nullptr, q16, T_);
    gemm16_kernel<0, false><<<GB, 128, 0, stream>>>(x, rs_x, kw_t, C_, V_,
                                                    nullptr, nullptr, nullptr, nullptr, k16, T_);
    gemm16_kernel<1, false><<<GB, 128, 0, stream>>>(x, rs_x, vw_t, C_, V_,
                                                    nullptr, nullptr, nullptr, nullptr, vT16, T_);

    // stage 2: fused decayed attention -> retrieved (f16, pre-scaled by mem_out_scale)
    attn_kernel<<<dim3(T_ / 16, B_), 128, 0, stream>>>(q16, k16, vT16, ret16, decay_l, mem_os);

    // stage 3: x1 = x + mem_scale * (retrieved @ o_w^T)
    gemm16_kernel<2, true><<<GB, 128, 0, stream>>>(ret16, nullptr, ow_t, V_, C_,
                                                   x, nullptr, mem_sc, x1, nullptr, T_);

    // stage 4: vals = gelu(rms(x1) @ W_r + bias) * op_out_scale  (f16)
    row_rsqrt<<<M_ / 8, 256, 0, stream>>>(x1, rs_x1, M_);
    gemm16_kernel<3, false><<<GB, 128, 0, stream>>>(x1, rs_x1, wr_t, C_, V_,
                                                    nullptr, bias, op_os, nullptr, vals16, T_);

    // stage 5: out = x1 + op_scale * (vals @ write_w^T)
    gemm16_kernel<2, true><<<GB, 128, 0, stream>>>(vals16, nullptr, ww_t, V_, C_,
                                                   x1, nullptr, op_sc, out, nullptr, T_);
}
